// CMDenNet_41171556499475
// MI455X (gfx1250) — matrix operand order repaired and checked by reading, compiled, not run on hardware
//
#include <hip/hip_runtime.h>

typedef __attribute__((ext_vector_type(16))) _Float16 v16h;
typedef __attribute__((ext_vector_type(8)))  _Float16 v8h;
typedef __attribute__((ext_vector_type(8)))  float    v8f;

#define ACT_NONE     0
#define ACT_GELU     1
#define ACT_SILU     2
#define ACT_SOFTPLUS 3

__device__ __forceinline__ float act_apply(float v, int act) {
  switch (act) {
    case ACT_GELU:     return 0.5f * v * (1.0f + erff(v * 0.7071067811865475f));
    case ACT_SILU:     return v / (1.0f + expf(-v));
    case ACT_SOFTPLUS: return (v > 20.0f) ? v : log1pf(expf(v));
    default:           return v;
  }
}

// ---------------------------------------------------------------------------
// WMMA GEMM, LDS-staged macro-tile version.
// Block = 128 threads (4 waves). Tile: 64 rows x 32 cols of C; wave w owns
// rows [w*16, w*16+16) and both 16-col sub-tiles (A fragment reused across
// two v_wmma per K-step). K advances 32 per step.
//   - Staging is two-phase: (1) issue ALL 24 clamped global loads into
//     registers with no consumers (loads batch in flight and overlap the
//     barrier wait), (2) after the WAR barrier, mask+convert+ds_store.
//     This avoids both select-to-branch predication and per-element
//     s_wait_loadcnt serialization.
//   - Fragment fetch = contiguous 16B ds_load_b128 runs per the ISA layout:
//       A lane(hi,lo): K in [8hi..8hi+7] and [16+8hi..23+8hi]  (two v8h)
//       B lane(hi,lo): K in [8hi..8hi+7] and [16+8hi..23+8hi]  (two v8h)
//   - EXEC is all-ones at every v_wmma (ISA requirement).
// ---------------------------------------------------------------------------
__global__ void __launch_bounds__(128)
k_gemm(const float* __restrict__ A, int lda,
       const float* __restrict__ W,
       const float* __restrict__ bias,
       const float* __restrict__ res, int ldr,
       float* __restrict__ C, int ldc,
       int M, int N, int K, int act)
{
  __shared__ alignas(16) _Float16 aLDS[64 * 32];   // [m][k]
  __shared__ alignas(16) _Float16 bLDS[32 * 32];   // [n][k] (transposed)
  const int tid  = threadIdx.x;
  const int wave = tid >> 5;
  const int lane = tid & 31;
  const int hi = lane >> 4;
  const int lo = lane & 15;
  const int m0 = blockIdx.x * 64;
  const int n0 = blockIdx.y * 32;
  const int Mc = M - 1, Kc = K - 1, Nc = N - 1;
  v8f acc0 = {};
  v8f acc1 = {};
  for (int k0 = 0; k0 < K; k0 += 32) {
    // ---- phase 1: issue all staging loads (no consumers yet)
    float av[16];
#pragma unroll
    for (int i = 0; i < 16; ++i) {
      int idx = i * 128 + tid;             // 0..2047 -> A[m0+idx/32][k0+idx%32]
      int gm = m0 + (idx >> 5), gk = k0 + (idx & 31);
      int cm = gm > Mc ? Mc : gm;
      int ck = gk > Kc ? Kc : gk;
      av[i] = A[(long)cm * lda + ck];      // clamped, always valid
    }
    float bv[8];
#pragma unroll
    for (int i = 0; i < 8; ++i) {
      int idx = i * 128 + tid;             // 0..1023 -> W[k0+idx/32][n0+idx%32]
      int gk = k0 + (idx >> 5), gn = n0 + (idx & 31);
      int ck = gk > Kc ? Kc : gk;
      int cn = gn > Nc ? Nc : gn;
      bv[i] = W[(long)ck * N + cn];
    }
    // prefetch next A panel (speculative-safe, no counters)
    {
      int gm = m0 + (tid >> 1);
      int cm = gm > Mc ? Mc : gm;
      int nk = k0 + 32;
      if (nk <= Kc)
        __builtin_prefetch(&A[(long)cm * lda + nk], 0, 1);   // global_prefetch_b8
    }
    __syncthreads();   // WAR: previous iteration's fragment reads are done
    // ---- phase 2: mask (recomputed, cheap), convert, store to LDS
#pragma unroll
    for (int i = 0; i < 16; ++i) {
      int idx = i * 128 + tid;
      int gm = m0 + (idx >> 5), gk = k0 + (idx & 31);
      float msk = (gm <= Mc && gk <= Kc) ? 1.0f : 0.0f;
      aLDS[idx] = (_Float16)(av[i] * msk);
    }
#pragma unroll
    for (int i = 0; i < 8; ++i) {
      int idx = i * 128 + tid;
      int kk = idx >> 5, nn = idx & 31;
      int gk = k0 + kk, gn = n0 + nn;
      float msk = (gk <= Kc && gn <= Nc) ? 1.0f : 0.0f;
      bLDS[nn * 32 + kk] = (_Float16)(bv[i] * msk);
    }
    __syncthreads();
    // ---- fragment fetch from LDS (16B-aligned ds_load_b128 runs)
    const _Float16* ap = &aLDS[(wave * 16 + lo) * 32 + 8 * hi];
    v8h a0 = *(const v8h*)ap;
    v8h a1 = *(const v8h*)(ap + 16);
    v16h af = __builtin_shufflevector(a0, a1, 0, 1, 2, 3, 4, 5, 6, 7,
                                      8, 9, 10, 11, 12, 13, 14, 15);
    const _Float16* bp0 = &bLDS[lo * 32 + 8 * hi];
    v8h b00 = *(const v8h*)bp0;
    v8h b01 = *(const v8h*)(bp0 + 16);
    v16h bf0 = __builtin_shufflevector(b00, b01, 0, 1, 2, 3, 4, 5, 6, 7,
                                       8, 9, 10, 11, 12, 13, 14, 15);
    const _Float16* bp1 = &bLDS[(16 + lo) * 32 + 8 * hi];
    v8h b10 = *(const v8h*)bp1;
    v8h b11 = *(const v8h*)(bp1 + 16);
    v16h bf1 = __builtin_shufflevector(b10, b11, 0, 1, 2, 3, 4, 5, 6, 7,
                                       8, 9, 10, 11, 12, 13, 14, 15);
    acc0 = __builtin_amdgcn_wmma_f32_16x16x32_f16(false, af, false, bf0,
                                                  (short)0, acc0, false, false);
    acc1 = __builtin_amdgcn_wmma_f32_16x16x32_f16(false, af, false, bf1,
                                                  (short)0, acc1, false, false);
  }
  // ---- epilogue: C/D layout VGPR r -> row r+8*hi, col lo
#pragma unroll
  for (int r = 0; r < 8; ++r) {
    int row = m0 + wave * 16 + r + 8 * hi;
    if (row < M) {
      int col0 = n0 + lo;
      if (col0 < N) {
        float v = acc0[r];
        if (bias) v += bias[col0];
        v = act_apply(v, act);
        if (res) v += res[(long)row * ldr + col0];
        C[(long)row * ldc + col0] = v;
      }
      int col1 = n0 + 16 + lo;
      if (col1 < N) {
        float v = acc1[r];
        if (bias) v += bias[col1];
        v = act_apply(v, act);
        if (res) v += res[(long)row * ldr + col1];
        C[(long)row * ldc + col1] = v;
      }
    }
  }
}

// ---------------------------------------------------------------------------
// Farthest-point sampling: one block per (b,t) chunk of 256 points (dim 6).
// ---------------------------------------------------------------------------
__global__ void k_fps(const float* __restrict__ blocks, float* __restrict__ out,
                      int per)
{
  __shared__ float sx[256], sy[256], sz[256], mind[256], rv[256];
  __shared__ int ri[256], sel[256];
  const int bt = blockIdx.x;
  const int tid = threadIdx.x;
  const float* base = blocks + (long)bt * 256 * 6;
  sx[tid] = base[tid * 6 + 0];
  sy[tid] = base[tid * 6 + 1];
  sz[tid] = base[tid * 6 + 2];
  if (tid == 0) sel[0] = 0;
  __syncthreads();
  {
    float dx = sx[tid] - sx[0], dy = sy[tid] - sy[0], dz = sz[tid] - sz[0];
    mind[tid] = dx * dx + dy * dy + dz * dz;
  }
  __syncthreads();
  for (int s = 1; s < per; ++s) {
    rv[tid] = mind[tid]; ri[tid] = tid;
    __syncthreads();
    for (int off = 128; off > 0; off >>= 1) {
      if (tid < off) {
        float ov = rv[tid + off]; int oi = ri[tid + off];
        if (ov > rv[tid] || (ov == rv[tid] && oi < ri[tid])) { rv[tid] = ov; ri[tid] = oi; }
      }
      __syncthreads();
    }
    int nxt = ri[0];
    if (tid == 0) sel[s] = nxt;
    float dx = sx[tid] - sx[nxt], dy = sy[tid] - sy[nxt], dz = sz[tid] - sz[nxt];
    mind[tid] = fminf(mind[tid], dx * dx + dy * dy + dz * dz);
    __syncthreads();
  }
  if (tid < per) {
    int src = sel[tid];
    float* o = out + ((long)bt * per + tid) * 6;
#pragma unroll
    for (int c = 0; c < 6; ++c) o[c] = base[src * 6 + c];
  }
}

// ---------------------------------------------------------------------------
// KNN: thread-per-point insertion top-32 (stable, matches top_k tie order).
// ---------------------------------------------------------------------------
__global__ void k_knn(const float* __restrict__ pts, int* __restrict__ knn, int N)
{
  long gid = (long)blockIdx.x * blockDim.x + threadIdx.x;
  if (gid >= 2L * N) return;
  int b = (int)(gid / N), i = (int)(gid % N);
  const float* pb = pts + (long)b * N * 6;
  float xi = pb[i * 6], yi = pb[i * 6 + 1], zi = pb[i * 6 + 2];
  float si = xi * xi + yi * yi + zi * zi;
  float bd[32]; int bi[32];
#pragma unroll
  for (int k = 0; k < 32; ++k) { bd[k] = 3.4e38f; bi[k] = 0; }
  for (int j = 0; j < N; ++j) {
    if (j == i) continue;
    float xj = pb[j * 6], yj = pb[j * 6 + 1], zj = pb[j * 6 + 2];
    float sj = xj * xj + yj * yj + zj * zj;
    float d = si + sj - 2.0f * (xi * xj + yi * yj + zi * zj);
    if (d < bd[31]) {
      int k = 31;
      while (k > 0 && bd[k - 1] > d) { bd[k] = bd[k - 1]; bi[k] = bi[k - 1]; --k; }
      bd[k] = d; bi[k] = j;
    }
  }
#pragma unroll
  for (int k = 0; k < 32; ++k) knn[gid * 32 + k] = bi[k];
}

__global__ void k_rel(const float* __restrict__ pts, const int* __restrict__ knn,
                      float* __restrict__ rel, int N)
{
  long i = (long)blockIdx.x * blockDim.x + threadIdx.x;
  if (i >= 2L * N * 32) return;
  int k = (int)(i & 31);
  long pt = i >> 5;
  int b = (int)(pt / N), p = (int)(pt % N);
  int j = knn[pt * 32 + k];
  const float* pb = pts + (long)b * N * 6;
  rel[i * 3 + 0] = pb[j * 6 + 0] - pb[p * 6 + 0];
  rel[i * 3 + 1] = pb[j * 6 + 1] - pb[p * 6 + 1];
  rel[i * 3 + 2] = pb[j * 6 + 2] - pb[p * 6 + 2];
}

__global__ void k_nrmvar(const float* __restrict__ pts, const int* __restrict__ knn,
                         float* __restrict__ nv, int N)
{
  long i = (long)blockIdx.x * blockDim.x + threadIdx.x;
  if (i >= 2L * N * 32) return;
  int k = (int)(i & 31);
  long pt = i >> 5;
  int b = (int)(pt / N), p = (int)(pt % N);
  int j = knn[pt * 32 + k];
  const float* pb = pts + (long)b * N * 6;
  float nxp = pb[p * 6 + 3], nyp = pb[p * 6 + 4], nzp = pb[p * 6 + 5];
  float nxj = pb[j * 6 + 3], nyj = pb[j * 6 + 4], nzj = pb[j * 6 + 5];
  nv[i * 4 + 0] = nxj;
  nv[i * 4 + 1] = nyj;
  nv[i * 4 + 2] = nzj;
  nv[i * 4 + 3] = fabsf(1.0f - (nxp * nxj + nyp * nyj + nzp * nzj));
}

__global__ void k_max32(const float* __restrict__ in, float* __restrict__ out, long rows)
{
  long i = (long)blockIdx.x * blockDim.x + threadIdx.x;
  if (i >= rows * 128) return;
  int c = (int)(i & 127);
  long r = i >> 7;
  float m = -3.4e38f;
  for (int k = 0; k < 32; ++k) m = fmaxf(m, in[((r << 5) + k) * 128 + c]);
  out[i] = m;
}

__global__ void k_concat2(const float* __restrict__ a, int ca,
                          const float* __restrict__ b, int cb,
                          float* __restrict__ o, long rows)
{
  int ct = ca + cb;
  long i = (long)blockIdx.x * blockDim.x + threadIdx.x;
  if (i >= rows * ct) return;
  long r = i / ct; int c = (int)(i % ct);
  o[i] = (c < ca) ? a[r * ca + c] : b[r * cb + (c - ca)];
}

// ---------------------------------------------------------------------------
// Z-order: per-batch min/max, morton codes, LDS bitonic sort of stable keys.
// ---------------------------------------------------------------------------
__global__ void k_minmax(const float* __restrict__ pts, float* __restrict__ mn,
                         float* __restrict__ mx, int N)
{
  __shared__ float smn[3][256], smx[3][256];
  int b = blockIdx.x, tid = threadIdx.x;
  float lmn[3] = {3.4e38f, 3.4e38f, 3.4e38f};
  float lmx[3] = {-3.4e38f, -3.4e38f, -3.4e38f};
  for (int i = tid; i < N; i += 256) {
#pragma unroll
    for (int ax = 0; ax < 3; ++ax) {
      float v = pts[((long)b * N + i) * 6 + ax];
      lmn[ax] = fminf(lmn[ax], v);
      lmx[ax] = fmaxf(lmx[ax], v);
    }
  }
#pragma unroll
  for (int ax = 0; ax < 3; ++ax) { smn[ax][tid] = lmn[ax]; smx[ax][tid] = lmx[ax]; }
  __syncthreads();
  for (int off = 128; off > 0; off >>= 1) {
    if (tid < off) {
#pragma unroll
      for (int ax = 0; ax < 3; ++ax) {
        smn[ax][tid] = fminf(smn[ax][tid], smn[ax][tid + off]);
        smx[ax][tid] = fmaxf(smx[ax][tid], smx[ax][tid + off]);
      }
    }
    __syncthreads();
  }
  if (tid == 0) {
#pragma unroll
    for (int ax = 0; ax < 3; ++ax) { mn[b * 3 + ax] = smn[ax][0]; mx[b * 3 + ax] = smx[ax][0]; }
  }
}

__global__ void k_code(const float* __restrict__ pts, const float* __restrict__ mn,
                       const float* __restrict__ mx, int* __restrict__ keys, int N)
{
  long i = (long)blockIdx.x * blockDim.x + threadIdx.x;
  if (i >= 2L * N) return;
  int b = (int)(i / N), p = (int)(i % N);
  int g[3];
#pragma unroll
  for (int ax = 0; ax < 3; ++ax) {
    float v = pts[i * 6 + ax];
    float t = (v - mn[b * 3 + ax]) / (mx[b * 3 + ax] - mn[b * 3 + ax] + 1e-9f) * 64.0f;
    int gi = (int)t;
    g[ax] = gi < 0 ? 0 : (gi > 63 ? 63 : gi);
  }
  int code = 0;
#pragma unroll
  for (int bit = 0; bit < 6; ++bit)
#pragma unroll
    for (int ax = 0; ax < 3; ++ax)
      code |= (((g[ax] >> bit) & 1) << (3 * bit + ax));
  keys[i] = (code << 11) | p;   // stable composite key (idx in low 11 bits)
}

__global__ void k_sort(const int* __restrict__ keys, int* __restrict__ order,
                       int* __restrict__ inv, int N)
{
  __shared__ int sk[2048];
  int b = blockIdx.x, tid = threadIdx.x, nt = blockDim.x;
  for (int i = tid; i < N; i += nt) sk[i] = keys[(long)b * N + i];
  __syncthreads();
  for (int k = 2; k <= N; k <<= 1) {
    for (int j = k >> 1; j > 0; j >>= 1) {
      for (int i = tid; i < N; i += nt) {
        int ixj = i ^ j;
        if (ixj > i) {
          int a = sk[i], c2 = sk[ixj];
          bool up = ((i & k) == 0);
          if ((up && a > c2) || (!up && a < c2)) { sk[i] = c2; sk[ixj] = a; }
        }
      }
      __syncthreads();
    }
  }
  for (int r = tid; r < N; r += nt) {
    int idx = sk[r] & 2047;
    order[(long)b * N + r] = idx;
    inv[(long)b * N + idx] = r;
  }
}

// ---------------------------------------------------------------------------
// Mamba block pieces
// ---------------------------------------------------------------------------
__global__ void k_gather(const float* __restrict__ feats, const int* __restrict__ order,
                         const float* __restrict__ oipb, const float* __restrict__ oipw,
                         float* __restrict__ seq, int N, int L)
{
  long i = (long)blockIdx.x * blockDim.x + threadIdx.x;
  if (i >= 2L * L * 128) return;
  int c = (int)(i & 127);
  long rest = i >> 7;
  int pos = (int)(rest % L);
  int b = (int)(rest / L);
  float v;
  if (pos == 0) v = oipb[c];
  else if (pos == L - 1) v = oipw[c] + oipb[c];
  else v = feats[((long)b * N + order[(long)b * N + pos - 1]) * 128 + c];
  seq[i] = v;
}

__global__ void k_ln(const float* __restrict__ x, const float* __restrict__ g,
                     const float* __restrict__ bb, float* __restrict__ y)
{
  __shared__ float red[128];
  long row = blockIdx.x;
  int c = threadIdx.x;
  float v = x[row * 128 + c];
  red[c] = v;
  __syncthreads();
  for (int off = 64; off > 0; off >>= 1) { if (c < off) red[c] += red[c + off]; __syncthreads(); }
  float mean = red[0] * (1.0f / 128.0f);
  __syncthreads();
  float d = v - mean;
  red[c] = d * d;
  __syncthreads();
  for (int off = 64; off > 0; off >>= 1) { if (c < off) red[c] += red[c + off]; __syncthreads(); }
  float var = red[0] * (1.0f / 128.0f);
  y[row * 128 + c] = d * rsqrtf(var + 1e-5f) * g[c] + bb[c];
}

__global__ void k_dwconv(const float* __restrict__ inp, const float* __restrict__ w,
                         const float* __restrict__ cb, float* __restrict__ xi, int L)
{
  long i = (long)blockIdx.x * blockDim.x + threadIdx.x;
  if (i >= 2L * L * 128) return;
  int c = (int)(i & 127);
  long rest = i >> 7;
  int t = (int)(rest % L);
  int b = (int)(rest / L);
  float acc = cb[c];
#pragma unroll
  for (int j = 0; j < 4; ++j) {
    int tt = t - 3 + j;
    if (tt >= 0) acc += inp[((long)b * L + tt) * 256 + c] * w[c * 4 + j];
  }
  xi[i] = acc / (1.0f + expf(-acc));   // silu
}

// Sequential selective scan: 1 lane per channel, 16-wide state in registers.
__global__ void k_ssm(const float* __restrict__ xi, const float* __restrict__ dbc,
                      const float* __restrict__ dt, const float* __restrict__ A_log,
                      const float* __restrict__ Dp, float* __restrict__ y,
                      int L, int backward, int accum)
{
  int b = blockIdx.x;
  int d = threadIdx.x;   // 128 channels
  float A[16];
#pragma unroll
  for (int s = 0; s < 16; ++s) A[s] = -expf(A_log[d * 16 + s]);
  float Dv = Dp[d];
  float h[16];
#pragma unroll
  for (int s = 0; s < 16; ++s) h[s] = 0.0f;
  for (int step = 0; step < L; ++step) {
    int t = backward ? (L - 1 - step) : step;
    long row = (long)b * L + t;
    float dtv = dt[row * 128 + d];
    float x = xi[row * 128 + d];
    float yv = 0.0f;
#pragma unroll
    for (int s = 0; s < 16; ++s) {
      float Bv = dbc[row * 40 + 8 + s];
      float Cv = dbc[row * 40 + 24 + s];
      h[s] = expf(dtv * A[s]) * h[s] + dtv * x * Bv;
      yv += h[s] * Cv;
    }
    yv += x * Dv;
    if (accum) y[row * 128 + d] += yv; else y[row * 128 + d] = yv;
  }
}

__global__ void k_gate(float* __restrict__ y, const float* __restrict__ inp, long rowsL)
{
  long i = (long)blockIdx.x * blockDim.x + threadIdx.x;
  if (i >= rowsL * 128) return;
  int c = (int)(i & 127);
  long r = i >> 7;
  float z = inp[r * 256 + 128 + c];
  y[i] *= z / (1.0f + expf(-z));
}

__global__ void k_scatter(const float* __restrict__ seq, const int* __restrict__ order,
                          float* __restrict__ feats, int N, int L)
{
  long i = (long)blockIdx.x * blockDim.x + threadIdx.x;
  if (i >= 2L * N * 128) return;
  int c = (int)(i & 127);
  long rest = i >> 7;
  int r = (int)(rest % N);
  int b = (int)(rest / N);
  feats[((long)b * N + order[(long)b * N + r]) * 128 + c] =
      seq[((long)b * L + r + 1) * 128 + c];
}

__global__ void k_pool(const float* __restrict__ feats, float* __restrict__ pooled, int N)
{
  int b = blockIdx.x;
  int c = threadIdx.x;   // 128
  float mx = -3.4e38f, sm = 0.0f;
  for (int i = 0; i < N; ++i) {
    float v = feats[((long)b * N + i) * 128 + c];
    mx = fmaxf(mx, v);
    sm += v;
  }
  pooled[b * 256 + c] = mx;
  pooled[b * 256 + 128 + c] = sm / (float)N;
}

__global__ void k_expand_add(float* __restrict__ dst, const float* __restrict__ src,
                             int n, int nsrc)
{
  long i = (long)blockIdx.x * blockDim.x + threadIdx.x;
  if (i >= 2L * n * 3) return;
  int c = (int)(i % 3);
  long rest = i / 3;
  int p = (int)(rest % n);
  int b = (int)(rest / n);
  int rep = n / nsrc;
  dst[((long)b * n + p) * 3 + c] += src[((long)b * nsrc + p / rep) * 3 + c];
}

// ---------------------------------------------------------------------------
// Host side
// ---------------------------------------------------------------------------
struct HDense { const float* b; const float* w; };
struct HMlp   { HDense l1, l2; };
struct HBlock {
  const float* A_log; const float* Dp; const float* conv_b; const float* conv_w;
  HDense dt; HDense in; const float* ln_b; const float* ln_g; HDense out;
  const float* xp_w;
};
struct HEnc {
  HBlock blk[4]; HDense in; const float* norm_b; const float* norm_g;
  HDense oip; HMlp out;
};

extern "C" void kernel_launch(void* const* d_in, const int* in_sizes, int n_in,
                              void* d_out, int out_size, void* d_ws, size_t ws_size,
                              hipStream_t stream)
{
  (void)in_sizes; (void)n_in; (void)out_size; (void)ws_size;
  const float* arch_blocks = (const float*)d_in[1];   // (2,16,256,6); d_in[0] unused

  // ---- params pytree flattened in JAX sorted-key order, starting at index 2
  int cur = 2;
  auto nf  = [&]() -> const float* { return (const float*)d_in[cur++]; };
  auto nD  = [&]() { HDense d; d.b = nf(); d.w = nf(); return d; };
  auto nM  = [&]() { HMlp m; m.l1 = nD(); m.l2 = nD(); return m; };
  auto nBk = [&]() {
    HBlock k; k.A_log = nf(); k.Dp = nf(); k.conv_b = nf(); k.conv_w = nf();
    k.dt = nD(); k.in = nD(); k.ln_b = nf(); k.ln_g = nf(); k.out = nD();
    k.xp_w = nf(); return k;
  };
  auto nE = [&]() {
    HEnc e; for (int i = 0; i < 4; ++i) e.blk[i] = nBk();
    e.in = nD(); e.norm_b = nf(); e.norm_g = nf(); e.oip = nD(); e.out = nM();
    return e;
  };
  HDense dec_high = nD(), dec_low = nD(), dec_mid = nD();       // 'dec': high,low,mid
  HEnc enc_high = nE(), enc_low = nE(), enc_mid = nE();          // sorted enc_* keys
  HMlp enh_coord = nM(), enh_normal = nM(), enh_out = nM();      // 'enh': coord,normal,out
  HMlp g1 = nM(), g2 = nM(), g3 = nM();

  // ---- workspace carve-out (sized for the HIGH encoder, N=2048)
  char* wp = (char*)d_ws;
  auto alloc = [&](size_t nelem) -> float* {
    float* p = (float*)wp;
    wp += ((nelem * sizeof(float)) + 255) & ~(size_t)255;
    return p;
  };
  float* s_sampled = alloc(2L * 2048 * 6);
  int*   s_knn     = (int*)alloc(2L * 2048 * 32);
  float* s_rel     = alloc(2L * 2048 * 32 * 3);
  float* s_nv      = alloc(2L * 2048 * 32 * 4);
  float* s_big0    = alloc(2L * 2048 * 32 * 128);
  float* s_big1    = alloc(2L * 2048 * 32 * 128);
  float* s_coord   = alloc(2L * 2048 * 128);
  float* s_normal  = alloc(2L * 2048 * 128);
  float* s_cat     = alloc(2L * 2048 * 256);
  float* s_enh     = alloc(2L * 2048 * 128);
  float* s_pts     = alloc(2L * 2048 * 134);
  float* s_feats   = alloc(2L * 2048 * 128);
  float* s_mn      = alloc(8);
  float* s_mx      = alloc(8);
  int*   s_keys    = (int*)alloc(2L * 2048);
  int*   s_order   = (int*)alloc(2L * 2048);
  int*   s_inv     = (int*)alloc(2L * 2048);
  const long Lmax = 2050;
  float* s_seq  = alloc(2L * Lmax * 128);
  float* s_lnb  = alloc(2L * Lmax * 128);
  float* s_inp  = alloc(2L * Lmax * 256);
  float* s_xi   = alloc(2L * Lmax * 128);
  float* s_dbc  = alloc(2L * Lmax * 40);
  float* s_dt   = alloc(2L * Lmax * 128);
  float* s_y    = alloc(2L * Lmax * 128);
  float* s_pool = alloc(2L * 256);
  float* s_h512 = alloc(2L * 512);
  float* s_gh   = alloc(2L * 512);
  float* s_f1   = alloc(2L * 512);
  float* s_f2   = alloc(2L * 512);
  float* s_f3   = alloc(2L * 512);

  auto ew = [](long total) { return dim3((unsigned)((total + 255) / 256)); };
  auto gemm = [&](const float* A, int lda, const float* W, const float* bias,
                  const float* res, int ldr, float* C, int ldc,
                  int M, int N, int K, int act) {
    dim3 g((unsigned)((M + 63) / 64), (unsigned)((N + 31) / 32));
    k_gemm<<<g, dim3(128), 0, stream>>>(A, lda, W, bias, res, ldr, C, ldc, M, N, K, act);
  };

  float* out = (float*)d_out;
  float* gf  = out + 21504;   // pl(3072) + pm(6144) + ph(12288)

  auto run_encode = [&](int N, const HEnc& E, int gfcol) {
    int per = N / 16;                                   // total // t
    k_fps<<<dim3(32), dim3(256), 0, stream>>>(arch_blocks, s_sampled, per);
    long BN = 2L * N;
    long Rn = BN * 32;
    // ---- enhancer
    k_knn<<<ew(BN), dim3(256), 0, stream>>>(s_sampled, s_knn, N);
    k_rel<<<ew(Rn), dim3(256), 0, stream>>>(s_sampled, s_knn, s_rel, N);
    gemm(s_rel, 3, enh_coord.l1.w, enh_coord.l1.b, nullptr, 0, s_big0, 128, (int)Rn, 128, 3, ACT_GELU);
    gemm(s_big0, 128, enh_coord.l2.w, enh_coord.l2.b, nullptr, 0, s_big1, 128, (int)Rn, 128, 128, ACT_NONE);
    k_max32<<<ew(BN * 128), dim3(256), 0, stream>>>(s_big1, s_coord, BN);
    k_nrmvar<<<ew(Rn), dim3(256), 0, stream>>>(s_sampled, s_knn, s_nv, N);
    gemm(s_nv, 4, enh_normal.l1.w, enh_normal.l1.b, nullptr, 0, s_big0, 128, (int)Rn, 128, 4, ACT_GELU);
    gemm(s_big0, 128, enh_normal.l2.w, enh_normal.l2.b, nullptr, 0, s_big1, 128, (int)Rn, 128, 128, ACT_NONE);
    k_max32<<<ew(BN * 128), dim3(256), 0, stream>>>(s_big1, s_normal, BN);
    k_concat2<<<ew(BN * 256), dim3(256), 0, stream>>>(s_coord, 128, s_normal, 128, s_cat, BN);
    gemm(s_cat, 256, enh_out.l1.w, enh_out.l1.b, nullptr, 0, s_big0, 128, (int)BN, 128, 256, ACT_GELU);
    gemm(s_big0, 128, enh_out.l2.w, enh_out.l2.b, nullptr, 0, s_enh, 128, (int)BN, 128, 128, ACT_NONE);
    // ---- encoder input + z-order
    k_concat2<<<ew(BN * 134), dim3(256), 0, stream>>>(s_sampled, 6, s_enh, 128, s_pts, BN);
    gemm(s_pts, 134, E.in.w, E.in.b, nullptr, 0, s_feats, 128, (int)BN, 128, 134, ACT_NONE);
    k_minmax<<<dim3(2), dim3(256), 0, stream>>>(s_sampled, s_mn, s_mx, N);
    k_code<<<ew(BN), dim3(256), 0, stream>>>(s_sampled, s_mn, s_mx, s_keys, N);
    k_sort<<<dim3(2), dim3(256), 0, stream>>>(s_keys, s_order, s_inv, N);
    // ---- 4 Mamba blocks
    int L = N + 2;
    long ML = 2L * L;
    for (int bI = 0; bI < 4; ++bI) {
      const HBlock& Bk = E.blk[bI];
      k_gather<<<ew(ML * 128), dim3(256), 0, stream>>>(s_feats, s_order, E.oip.b, E.oip.w, s_seq, N, L);
      k_ln<<<dim3((unsigned)ML), dim3(128), 0, stream>>>(s_seq, Bk.ln_g, Bk.ln_b, s_lnb);
      gemm(s_lnb, 128, Bk.in.w, Bk.in.b, nullptr, 0, s_inp, 256, (int)ML, 256, 128, ACT_NONE);
      k_dwconv<<<ew(ML * 128), dim3(256), 0, stream>>>(s_inp, Bk.conv_w, Bk.conv_b, s_xi, L);
      gemm(s_xi, 128, Bk.xp_w, nullptr, nullptr, 0, s_dbc, 40, (int)ML, 40, 128, ACT_NONE);
      gemm(s_dbc, 40, Bk.dt.w, Bk.dt.b, nullptr, 0, s_dt, 128, (int)ML, 128, 8, ACT_SOFTPLUS);
      k_ssm<<<dim3(2), dim3(128), 0, stream>>>(s_xi, s_dbc, s_dt, Bk.A_log, Bk.Dp, s_y, L, 0, 0);
      k_ssm<<<dim3(2), dim3(128), 0, stream>>>(s_xi, s_dbc, s_dt, Bk.A_log, Bk.Dp, s_y, L, 1, 1);
      k_gate<<<ew(ML * 128), dim3(256), 0, stream>>>(s_y, s_inp, ML);
      gemm(s_y, 128, Bk.out.w, Bk.out.b, s_seq, 128, s_seq, 128, (int)ML, 128, 128, ACT_NONE);
      k_scatter<<<ew(BN * 128), dim3(256), 0, stream>>>(s_seq, s_order, s_feats, N, L);
    }
    // ---- head
    k_ln<<<dim3((unsigned)BN), dim3(128), 0, stream>>>(s_feats, E.norm_g, E.norm_b, s_feats);
    k_pool<<<dim3(2), dim3(128), 0, stream>>>(s_feats, s_pool, N);
    gemm(s_pool, 256, E.out.l1.w, E.out.l1.b, nullptr, 0, s_h512, 512, 2, 512, 256, ACT_GELU);
    gemm(s_h512, 512, E.out.l2.w, E.out.l2.b, nullptr, 0, gf + gfcol, 1536, 2, 512, 512, ACT_NONE);
  };

  run_encode(512,  enc_low,  0);
  run_encode(1024, enc_mid,  512);
  run_encode(2048, enc_high, 1024);

  // ---- global MLPs
  gemm(gf, 1536, g1.l1.w, g1.l1.b, nullptr, 0, s_gh, 512, 2, 512, 1536, ACT_GELU);
  gemm(s_gh, 512, g1.l2.w, g1.l2.b, nullptr, 0, s_f1, 512, 2, 512, 512, ACT_NONE);
  gemm(gf, 1536, g2.l1.w, g2.l1.b, nullptr, 0, s_gh, 512, 2, 512, 1536, ACT_GELU);
  gemm(s_gh, 512, g2.l2.w, g2.l2.b, nullptr, 0, s_f2, 512, 2, 512, 512, ACT_NONE);
  gemm(gf, 1536, g3.l1.w, g3.l1.b, nullptr, 0, s_gh, 512, 2, 512, 1536, ACT_GELU);
  gemm(s_gh, 512, g3.l2.w, g3.l2.b, nullptr, 0, s_f3, 512, 2, 512, 512, ACT_NONE);

  // ---- decoders: pl @0, pm @3072, ph @9216
  gemm(s_f3, 512, dec_low.w,  dec_low.b,  nullptr, 0, out,        1536, 2, 1536, 512, ACT_NONE);
  gemm(s_f2, 512, dec_mid.w,  dec_mid.b,  nullptr, 0, out + 3072, 3072, 2, 3072, 512, ACT_NONE);
  k_expand_add<<<ew(2L * 1024 * 3), dim3(256), 0, stream>>>(out + 3072, out, 1024, 512);
  gemm(s_f1, 512, dec_high.w, dec_high.b, nullptr, 0, out + 9216, 6144, 2, 6144, 512, ACT_NONE);
  k_expand_add<<<ew(2L * 2048 * 3), dim3(256), 0, stream>>>(out + 9216, out + 3072, 2048, 1024);
}
